// Struct2SeqGCN_52862457480030
// MI455X (gfx1250) — matrix operand order repaired and checked by reading, not compiled, not run on hardware
//
#include <hip/hip_runtime.h>
#include <hip/hip_bf16.h>
#include <math.h>

// ---------------------------------------------------------------------------
// Problem constants (from reference)
// ---------------------------------------------------------------------------
#define N_NODES 20000
#define N_EDGES 320000
#define F_IN    6
#define HID     128
#define N_CLS   21
#define N_LAYERS 4
#define NUM_G   16
#define RBF_STEP (8.0f / 15.0f)
#define RBF_COEFF (-1.7578125f)       // -0.5/(8/15)^2
#define EPS 1e-5f

typedef __attribute__((ext_vector_type(16))) __bf16 v16bf;
typedef __attribute__((ext_vector_type(8)))  float  v8f;
typedef __attribute__((ext_vector_type(8)))  __bf16 bf8;
typedef __attribute__((ext_vector_type(4)))  __bf16 bf4;

// ---------------------------------------------------------------------------
// 0) Pack Wf/Ws (per layer) into bf16 fragments laid out exactly as the WMMA
//    B operand lanes consume them: for (ntile, kstep) each lane owns column
//    j = ntile*16 + lane%16 and the contiguous K range (lane/16)*16 .. +15.
//    Wcat columns: [Wf_dst(0:128) | Wf_src | Ws_dst | Ws_src] -> 512 cols.
// ---------------------------------------------------------------------------
__global__ void pack_weights_kernel(const float* __restrict__ Wf,
                                    const float* __restrict__ Ws,
                                    __bf16* __restrict__ wpack) {
    int idx = blockIdx.x * blockDim.x + threadIdx.x;   // 4 * 65536 threads
    int l    = idx >> 16;
    int rem  = idx & 65535;
    int ntile = rem >> 11;          // /2048
    int rem2  = rem & 2047;
    int kstep = rem2 >> 9;          // /512
    int t     = rem2 & 511;
    int lane  = t >> 4;
    int i     = t & 15;
    int j = ntile * 16 + (lane & 15);
    int k = kstep * 32 + ((lane >> 4) << 4) + i;
    const float* WfL = Wf + l * 272 * HID;
    const float* WsL = Ws + l * 272 * HID;
    float v;
    if (j < 128)      v = WfL[k * HID + j];                 // A_f  (dst part of Wf)
    else if (j < 256) v = WfL[(128 + k) * HID + (j - 128)]; // B_f  (src part)
    else if (j < 384) v = WsL[k * HID + (j - 256)];         // A_s
    else              v = WsL[(128 + k) * HID + (j - 384)]; // B_s
    wpack[idx] = (__bf16)v;
}

// ---------------------------------------------------------------------------
// 1) Node embedding: h = x @ W_node + b_node   (F_IN=6 -> plain VALU)
// ---------------------------------------------------------------------------
__global__ void embed_kernel(const float* __restrict__ x,
                             const float* __restrict__ Wn,
                             const float* __restrict__ bn,
                             float* __restrict__ h,
                             __bf16* __restrict__ hbf) {
    int idx = blockIdx.x * blockDim.x + threadIdx.x;   // N_NODES*HID
    int n = idx >> 7, c = idx & 127;
    float acc = bn[c];
#pragma unroll
    for (int k = 0; k < F_IN; ++k) acc = fmaf(x[n * F_IN + k], Wn[k * HID + c], acc);
    h[idx]   = acc;
    hbf[idx] = (__bf16)acc;
}

// ---------------------------------------------------------------------------
// 2) Node GEMM via WMMA: P[N,512] = hbf[N,128] @ Wcat[128,512]
//    One wave computes a 16x64 strip: 4 N-tiles sharing one A fragment per
//    K-step -> 16 x v_wmma_f32_16x16x32_bf16 per wave, A loaded once/kstep.
//    A k order used here (matches wpack B): lane half -> contiguous K {base..base+15}.
// ---------------------------------------------------------------------------
__global__ void node_gemm_wmma_kernel(const __bf16* __restrict__ hbf,
                                      const __bf16* __restrict__ wpack_layer,
                                      float* __restrict__ P) {
    int wave = (blockIdx.x * blockDim.x + threadIdx.x) >> 5;  // 10000 waves
    int lane = threadIdx.x & 31;
    int mtile = wave >> 3;          // 1250 M tiles
    int ng    = wave & 7;           // N-tile group: tiles ng*4 .. ng*4+3
    int m0 = mtile << 4;

    int row   = m0 + (lane & 15);
    int abase = (lane >> 4) << 4;                   // 0 or 16
    const __bf16* ap = hbf + (size_t)row * HID;
    const __bf16* bp = wpack_layer + (size_t)(ng * 4) * 2048 + lane * 16;

    v8f acc[4];
#pragma unroll
    for (int t = 0; t < 4; ++t) acc[t] = (v8f){0.f,0.f,0.f,0.f,0.f,0.f,0.f,0.f};

#pragma unroll
    for (int ks = 0; ks < 4; ++ks) {
        bf8 alo = *(const bf8*)(ap + ks * 32 + abase);
        bf8 ahi = *(const bf8*)(ap + ks * 32 + abase + 8);
        v16bf a;
#pragma unroll
        for (int i = 0; i < 8; ++i) { a[i] = alo[i]; a[8 + i] = ahi[i]; }
#pragma unroll
        for (int t = 0; t < 4; ++t) {
            v16bf b = *(const v16bf*)(bp + (size_t)t * 2048 + ks * 512);
            acc[t] = __builtin_amdgcn_wmma_f32_16x16x32_bf16(
                /*neg_a=*/false, a, /*neg_b=*/false, b,
                /*c_mod=*/(short)0, acc[t], /*reuse_a=*/false, /*reuse_b=*/false);
        }
    }
    // D layout: VGPR r, lane l -> (M = m0 + r + 8*(l/16), N = j0 + l%16)
    int orow = m0 + ((lane >> 4) << 3);
#pragma unroll
    for (int t = 0; t < 4; ++t) {
        int j0 = (ng * 4 + t) << 4;
        float* op = P + (size_t)orow * 512 + j0 + (lane & 15);
#pragma unroll
        for (int r = 0; r < 8; ++r) op[(size_t)r * 512] = acc[t][r];
    }
}

// ---------------------------------------------------------------------------
// 3) Edge messages + segment_sum scatter.
//    One wave per edge, 4 channels per lane.
//    f = P[dst][0:128] + P[src][128:256] + e@Wf_rbf + bf   (same for s)
//    msg = sigmoid(f) * softplus(s);  atomicAdd into agg[dst].
// ---------------------------------------------------------------------------
__global__ void edge_msg_kernel(const float* __restrict__ P,
                                const int* __restrict__ ei,
                                const float* __restrict__ dist,
                                const float* __restrict__ WfL,   // layer base (272x128)
                                const float* __restrict__ WsL,
                                const float* __restrict__ bfL,
                                const float* __restrict__ bsL,
                                float* __restrict__ agg) {
    __shared__ float sWf[NUM_G * HID];
    __shared__ float sWs[NUM_G * HID];
    __shared__ float sbf[HID];
    __shared__ float sbs[HID];
    int tid = threadIdx.x;
    for (int i = tid; i < NUM_G * HID; i += 256) {
        sWf[i] = WfL[256 * HID + i];   // rbf rows 256..271 are contiguous
        sWs[i] = WsL[256 * HID + i];
    }
    if (tid < HID) { sbf[tid] = bfL[tid]; sbs[tid] = bsL[tid]; }
    __syncthreads();

    int wave = tid >> 5, lane = tid & 31;
    int e = blockIdx.x * 8 + wave;
    if (e >= N_EDGES) return;
    int src = ei[e];
    int dst = ei[N_EDGES + e];
    float d = dist[e];
    int c = lane * 4;

    const float* pd = P + (size_t)dst * 512;
    const float* ps = P + (size_t)src * 512;
    float4 fa = *(const float4*)(pd + c);
    float4 fb = *(const float4*)(ps + 128 + c);
    float4 sa = *(const float4*)(pd + 256 + c);
    float4 sb = *(const float4*)(ps + 384 + c);

    // Pull the atomic-destination line toward L2 while we burn VALU cycles on
    // the transcendental tail below (emits global_prefetch_b8 on gfx1250).
    float* ag = agg + (size_t)dst * HID + c;
    __builtin_prefetch((const void*)ag, 1, 1);

    float f[4] = { fa.x + fb.x + sbf[c + 0], fa.y + fb.y + sbf[c + 1],
                   fa.z + fb.z + sbf[c + 2], fa.w + fb.w + sbf[c + 3] };
    float s[4] = { sa.x + sb.x + sbs[c + 0], sa.y + sb.y + sbs[c + 1],
                   sa.z + sb.z + sbs[c + 2], sa.w + sb.w + sbs[c + 3] };

#pragma unroll
    for (int k = 0; k < NUM_G; ++k) {
        float dk = d - RBF_STEP * (float)k;
        float ek = __expf(RBF_COEFF * dk * dk);
        const float* wf = sWf + k * HID + c;
        const float* ws = sWs + k * HID + c;
#pragma unroll
        for (int i = 0; i < 4; ++i) {
            f[i] = fmaf(ek, wf[i], f[i]);
            s[i] = fmaf(ek, ws[i], s[i]);
        }
    }

#pragma unroll
    for (int i = 0; i < 4; ++i) {
        float sig = 1.0f / (1.0f + __expf(-f[i]));
        float sp  = (s[i] > 20.0f) ? s[i] : log1pf(__expf(s[i]));
        atomicAdd(ag + i, sig * sp);
    }
}

// ---------------------------------------------------------------------------
// 4) BatchNorm stats: per-channel sum / sumsq over all nodes.
// ---------------------------------------------------------------------------
__global__ void bn_stats_kernel(const float* __restrict__ agg,
                                float* __restrict__ stats) {
    __shared__ float s0[256];
    __shared__ float s1[256];
    int tid = threadIdx.x;
    int c = tid & 127, half = tid >> 7;
    float sum = 0.f, sq = 0.f;
    int r0 = blockIdx.x * 128 + half;
    for (int t = 0; t < 64; ++t) {
        int r = r0 + 2 * t;
        if (r < N_NODES) {
            float v = agg[(size_t)r * HID + c];
            sum += v; sq = fmaf(v, v, sq);
        }
    }
    s0[tid] = sum; s1[tid] = sq;
    __syncthreads();
    if (tid < HID) {
        atomicAdd(stats + c,       s0[tid] + s0[tid + 128]);
        atomicAdd(stats + HID + c, s1[tid] + s1[tid + 128]);
    }
}

__global__ void bn_finalize_kernel(const float* __restrict__ stats,
                                   float* __restrict__ murstd) {
    int c = threadIdx.x;   // 128 threads
    float mean = stats[c] * (1.0f / (float)N_NODES);
    float var  = stats[HID + c] * (1.0f / (float)N_NODES) - mean * mean;
    murstd[c]       = mean;
    murstd[HID + c] = rsqrtf(var + EPS);
}

// ---------------------------------------------------------------------------
// 5) Fused node update: BN-apply + residual + LayerNorm + ReLU + residual.
//    One wave per node; wave32 __shfl_xor row reductions.
// ---------------------------------------------------------------------------
__global__ void node_update_kernel(const float* __restrict__ agg,
                                   float* __restrict__ h,
                                   __bf16* __restrict__ hbf,
                                   const float* __restrict__ murstd,
                                   const float* __restrict__ bng,
                                   const float* __restrict__ bnb,
                                   const float* __restrict__ lng,
                                   const float* __restrict__ lnb) {
    int wave = threadIdx.x >> 5, lane = threadIdx.x & 31;
    int node = blockIdx.x * 8 + wave;
    int c = lane * 4;
    float4 a4 = *(const float4*)(agg + (size_t)node * HID + c);
    float4 h4 = *(const float4*)(h   + (size_t)node * HID + c);
    float av[4] = { a4.x, a4.y, a4.z, a4.w };
    float hh[4] = { h4.x, h4.y, h4.z, h4.w };
    float co[4];
    float s = 0.f, q = 0.f;
#pragma unroll
    for (int i = 0; i < 4; ++i) {
        float a = (av[i] - murstd[c + i]) * murstd[HID + c + i] * bng[c + i] + bnb[c + i];
        co[i] = a + hh[i];
        s += co[i]; q = fmaf(co[i], co[i], q);
    }
#pragma unroll
    for (int off = 16; off >= 1; off >>= 1) {
        s += __shfl_xor(s, off);
        q += __shfl_xor(q, off);
    }
    float m = s * (1.0f / (float)HID);
    float v = q * (1.0f / (float)HID) - m * m;
    float r = rsqrtf(v + EPS);
    float hn[4];
#pragma unroll
    for (int i = 0; i < 4; ++i) {
        float ln = (co[i] - m) * r * lng[c + i] + lnb[c + i];
        hn[i] = fmaxf(ln, 0.0f) + hh[i];
    }
    *(float4*)(h + (size_t)node * HID + c) = make_float4(hn[0], hn[1], hn[2], hn[3]);
    bf4 hb;
#pragma unroll
    for (int i = 0; i < 4; ++i) hb[i] = (__bf16)hn[i];
    *(bf4*)(hbf + (size_t)node * HID + c) = hb;
}

// ---------------------------------------------------------------------------
// 6) Final LayerNorm + FC [128 -> 21]. One wave per node, LDS row staging.
// ---------------------------------------------------------------------------
__global__ void final_out_kernel(const float* __restrict__ h,
                                 const float* __restrict__ lng,
                                 const float* __restrict__ lnb,
                                 const float* __restrict__ Wfc,
                                 const float* __restrict__ bfc,
                                 float* __restrict__ out) {
    __shared__ float lds[8][HID];
    int wave = threadIdx.x >> 5, lane = threadIdx.x & 31;
    int node = blockIdx.x * 8 + wave;
    int c = lane * 4;
    float4 h4 = *(const float4*)(h + (size_t)node * HID + c);
    float hv[4] = { h4.x, h4.y, h4.z, h4.w };
    float s = 0.f, q = 0.f;
#pragma unroll
    for (int i = 0; i < 4; ++i) { s += hv[i]; q = fmaf(hv[i], hv[i], q); }
#pragma unroll
    for (int off = 16; off >= 1; off >>= 1) {
        s += __shfl_xor(s, off);
        q += __shfl_xor(q, off);
    }
    float m = s * (1.0f / (float)HID);
    float v = q * (1.0f / (float)HID) - m * m;
    float r = rsqrtf(v + EPS);
#pragma unroll
    for (int i = 0; i < 4; ++i)
        lds[wave][c + i] = (hv[i] - m) * r * lng[c + i] + lnb[c + i];
    __syncthreads();
    if (lane < N_CLS) {
        float acc = bfc[lane];
        for (int k = 0; k < HID; ++k)
            acc = fmaf(lds[wave][k], Wfc[k * N_CLS + lane], acc);
        out[(size_t)node * N_CLS + lane] = acc;
    }
}

// ---------------------------------------------------------------------------
// Host: orchestrate on `stream` (graph-capture safe: kernels + memsetAsync).
// ---------------------------------------------------------------------------
extern "C" void kernel_launch(void* const* d_in, const int* in_sizes, int n_in,
                              void* d_out, int out_size, void* d_ws, size_t ws_size,
                              hipStream_t stream) {
    const float* x        = (const float*)d_in[0];
    const int*   ei       = (const int*)  d_in[1];
    const float* dist     = (const float*)d_in[2];
    const float* W_node   = (const float*)d_in[3];
    const float* b_node   = (const float*)d_in[4];
    const float* Wf       = (const float*)d_in[5];
    const float* bf       = (const float*)d_in[6];
    const float* Ws       = (const float*)d_in[7];
    const float* bs       = (const float*)d_in[8];
    const float* bn_g     = (const float*)d_in[9];
    const float* bn_b     = (const float*)d_in[10];
    const float* ln_g     = (const float*)d_in[11];
    const float* ln_b     = (const float*)d_in[12];
    const float* lnout_g  = (const float*)d_in[13];
    const float* lnout_b  = (const float*)d_in[14];
    const float* W_fc     = (const float*)d_in[15];
    const float* b_fc     = (const float*)d_in[16];
    float* out = (float*)d_out;

    // Workspace layout (all offsets multiples of 256 B)
    char* ws = (char*)d_ws;
    __bf16* hbf    = (__bf16*)(ws + 0);                       //  5,120,000 B
    __bf16* wpack  = (__bf16*)(ws + 5120000);                 //    524,288 B
    float*  P      = (float*) (ws + 5644288);                 // 40,960,000 B
    float*  agg    = (float*) (ws + 46604288);                // 10,240,000 B
    float*  h      = (float*) (ws + 56844288);                // 10,240,000 B
    float*  stats  = (float*) (ws + 67084288);                //      1,024 B
    float*  murstd = (float*) (ws + 67085312);                //      1,024 B
    (void)ws_size; (void)n_in; (void)in_sizes; (void)out_size;

    // 0) pack all layer weights to bf16 WMMA-B fragments
    pack_weights_kernel<<<(N_LAYERS * 65536) / 256, 256, 0, stream>>>(Wf, Ws, wpack);

    // 1) node embedding
    embed_kernel<<<(N_NODES * HID) / 256, 256, 0, stream>>>(x, W_node, b_node, h, hbf);

    for (int l = 0; l < N_LAYERS; ++l) {
        const float* WfL = Wf + (size_t)l * 272 * HID;
        const float* WsL = Ws + (size_t)l * 272 * HID;

        // 2) P = hbf @ Wcat via WMMA (1250 M-tiles x 8 N-groups = 10000 waves)
        node_gemm_wmma_kernel<<<1250, 256, 0, stream>>>(hbf, wpack + (size_t)l * 65536, P);

        // 3) messages + scatter
        hipMemsetAsync(agg, 0, (size_t)N_NODES * HID * sizeof(float), stream);
        edge_msg_kernel<<<N_EDGES / 8, 256, 0, stream>>>(
            P, ei, dist, WfL, WsL, bf + l * HID, bs + l * HID, agg);

        // 4) BatchNorm batch stats
        hipMemsetAsync(stats, 0, 2 * HID * sizeof(float), stream);
        bn_stats_kernel<<<(N_NODES + 127) / 128, 256, 0, stream>>>(agg, stats);
        bn_finalize_kernel<<<1, HID, 0, stream>>>(stats, murstd);

        // 5) fused BN + residual + LayerNorm + ReLU + residual
        node_update_kernel<<<N_NODES / 8, 256, 0, stream>>>(
            agg, h, hbf, murstd,
            bn_g + l * HID, bn_b + l * HID, ln_g + l * HID, ln_b + l * HID);
    }

    // 6) final LayerNorm + FC
    final_out_kernel<<<N_NODES / 8, 256, 0, stream>>>(h, lnout_g, lnout_b, W_fc, b_fc, out);
}
